// Attention_43963285242632
// MI455X (gfx1250) — matrix operand order repaired and checked by reading, not compiled, not run on hardware
//
#include <hip/hip_runtime.h>
#include <hip/hip_bf16.h>

typedef __bf16 bf16;
typedef __attribute__((ext_vector_type(4)))  bf16  bf16x4;
typedef __attribute__((ext_vector_type(8)))  bf16  bf16x8;
typedef __attribute__((ext_vector_type(16))) bf16  bf16x16;
typedef __attribute__((ext_vector_type(4)))  float f32x4;
typedef __attribute__((ext_vector_type(8)))  float f32x8;

#define B_   2
#define L_   2048
#define D_   2048
#define H_   16
#define KV_  4
#define HD_  64
#define NH_  (H_ + 2 * KV_)   /* 24 heads in qkv   */
#define E_   (NH_ * HD_)      /* 1536              */
#define O_   (H_ * HD_)       /* 1024              */

static __device__ __forceinline__ bf16x16 cat8(bf16x8 lo, bf16x8 hi) {
  return __builtin_shufflevector(lo, hi, 0,1,2,3,4,5,6,7,8,9,10,11,12,13,14,15);
}

// 16-bit A-matrix 16x32 fragment (ISA 7.12.2): lane m holds row m,
// half 0 -> K[0..8)+K[16..24), half 1 -> K[8..16)+K[24..32)
static __device__ __forceinline__ bf16x16 load_fragA(const bf16* rowbase, int lhalf) {
  bf16x8 lo = *(const bf16x8*)(rowbase + lhalf * 8);
  bf16x8 hi = *(const bf16x8*)(rowbase + lhalf * 8 + 16);
  return cat8(lo, hi);
}

// 16-bit B-matrix 32x16 fragment: lane n holds column n (== row n of B^T),
// half 0 -> K[0..8)+K[16..24), half 1 -> K[8..16)+K[24..32)
static __device__ __forceinline__ bf16x16 load_fragB(const bf16* colbase, int lhalf) {
  bf16x8 lo = *(const bf16x8*)(colbase + lhalf * 8);
  bf16x8 hi = *(const bf16x8*)(colbase + lhalf * 8 + 16);
  return cat8(lo, hi);
}

static __device__ __forceinline__ f32x8 wmma_bf16(bf16x16 a, bf16x16 b, f32x8 c) {
  return __builtin_amdgcn_wmma_f32_16x16x32_bf16(false, a, false, b, (short)0, c,
                                                 false, false);
}

// gfx1250 async global->LDS copy, 16 bytes per lane (tracked by ASYNCcnt).
// LDS operand is the low-32-bit LDS offset (generic LDS aperture: addr[31:0]).
static __device__ __forceinline__ void async_copy_b128(const bf16* gsrc, bf16* ldst) {
  unsigned lo = (unsigned)(unsigned long long)ldst;
  asm volatile("global_load_async_to_lds_b128 %0, %1, off"
               :: "v"(lo), "v"(gsrc) : "memory");
}
static __device__ __forceinline__ void wait_async0() {
  asm volatile("s_wait_asynccnt 0x0" ::: "memory");
}

// 16-lane butterfly max entirely in VALU via DPP16 (no ds_bpermute):
// quad_perm(1,0,3,2)=xor1, quad_perm(2,3,0,1)=xor2, row_half_mirror, row_mirror
template <int CTRL>
static __device__ __forceinline__ float dpp_f(float x) {
  int r = __builtin_amdgcn_update_dpp(0, __builtin_bit_cast(int, x),
                                      CTRL, 0xf, 0xf, true);
  return __builtin_bit_cast(float, r);
}
static __device__ __forceinline__ float redmax16(float v) {
  v = fmaxf(v, dpp_f<0xB1>(v));    // quad_perm [1,0,3,2]
  v = fmaxf(v, dpp_f<0x4E>(v));    // quad_perm [2,3,0,1]
  v = fmaxf(v, dpp_f<0x141>(v));   // row_half_mirror
  v = fmaxf(v, dpp_f<0x140>(v));   // row_mirror
  return v;
}

/* ---------------- fp32 -> bf16 conversion (vectorized x4) ---------------- */
__global__ void cvt_f32_bf16(const float* __restrict__ in, bf16* __restrict__ out, int n4) {
  int i = blockIdx.x * blockDim.x + threadIdx.x;
  if (i < n4) {
    f32x4 v = ((const f32x4*)in)[i];
    ((bf16x4*)out)[i] = __builtin_convertvector(v, bf16x4);
  }
}

/* ---------------- C = A(MxK) * B(NxK)^T, bf16 in / f32 out ----------------
   Compile-time N,K so the epilogue folds into immediate store offsets.
   Block tile 128x128, K step 32, double-buffered async-to-LDS staging,
   8 waves (2x4), each wave 64x32 via 4x2 WMMA tiles. */
template <int N, int K>
__global__ __launch_bounds__(256) void gemm_nt(
    const bf16* __restrict__ A, const bf16* __restrict__ Bm, float* __restrict__ C) {
  constexpr int LDSS = 40;  // halves; pad to kill bank conflicts
  __shared__ bf16 As[2][128 * LDSS];
  __shared__ bf16 Bs[2][128 * LDSS];

  const int tid  = threadIdx.x;
  const int lane = tid & 31;
  const int wave = tid >> 5;
  const int lrow = lane & 15;
  const int lhalf = lane >> 4;
  const int wm = wave >> 2;                      // 0..1
  const int wn = wave & 3;                       // 0..3
  const long bm = (long)blockIdx.x * 128;
  const long bn = (long)blockIdx.y * 128;
  const int srow = tid >> 2;                     // 0..63
  const int scol = (tid & 3) * 8;                // 0,8,16,24

  const bf16* ga0 = A  + (bm + srow)      * K + scol;
  const bf16* ga1 = A  + (bm + srow + 64) * K + scol;
  const bf16* gb0 = Bm + (bn + srow)      * K + scol;
  const bf16* gb1 = Bm + (bn + srow + 64) * K + scol;

  auto stage = [&](int kt, int buf) {
    async_copy_b128(ga0 + kt, &As[buf][(srow     ) * LDSS + scol]);
    async_copy_b128(ga1 + kt, &As[buf][(srow + 64) * LDSS + scol]);
    async_copy_b128(gb0 + kt, &Bs[buf][(srow     ) * LDSS + scol]);
    async_copy_b128(gb1 + kt, &Bs[buf][(srow + 64) * LDSS + scol]);
  };

  f32x8 acc[4][2];
  f32x8 zero = {0.f,0.f,0.f,0.f,0.f,0.f,0.f,0.f};
#pragma unroll
  for (int mi = 0; mi < 4; ++mi)
#pragma unroll
    for (int ni = 0; ni < 2; ++ni) acc[mi][ni] = zero;

  constexpr int T = K / 32;
  stage(0, 0);
  for (int t = 0; t < T; ++t) {
    const int cur = t & 1;
    wait_async0();          // my async writes for tile t have landed
    __syncthreads();        // everyone's have landed; tile t-1 reads done
    if (t + 1 < T) stage((t + 1) * 32, cur ^ 1);  // overlap with compute

    bf16x16 af[4], bfr[2];
#pragma unroll
    for (int mi = 0; mi < 4; ++mi)
      af[mi] = load_fragA(&As[cur][(wm * 64 + mi * 16 + lrow) * LDSS], lhalf);
#pragma unroll
    for (int ni = 0; ni < 2; ++ni)
      bfr[ni] = load_fragB(&Bs[cur][(wn * 32 + ni * 16 + lrow) * LDSS], lhalf);

#pragma unroll
    for (int mi = 0; mi < 4; ++mi)
#pragma unroll
      for (int ni = 0; ni < 2; ++ni)
        acc[mi][ni] = wmma_bf16(af[mi], bfr[ni], acc[mi][ni]);
  }

  // C layout: VGPR i -> M = i + 8*lhalf; N = lrow within 16-wide tile.
  // One base per 16-row strip; i*N and ni*16 are compile-time store offsets.
#pragma unroll
  for (int mi = 0; mi < 4; ++mi) {
    float* cb = C + (bm + wm * 64 + mi * 16 + 8 * lhalf) * N
                  + (bn + wn * 32 + lrow);
#pragma unroll
    for (int ni = 0; ni < 2; ++ni)
#pragma unroll
      for (int i = 0; i < 8; ++i)
        cb[i * N + ni * 16] = acc[mi][ni][i];
  }
}

/* ---------------- RMSNorm + RoPE (+ fold 1/sqrt(HD) into Q), V transpose ----------------
   One wave per (b,l,head) row of 64; lane t handles elements t and t+32 (a RoPE pair). */
__global__ __launch_bounds__(256) void qkv_post(
    const float* __restrict__ qkv, const float* __restrict__ qw, const float* __restrict__ kw,
    bf16* __restrict__ qb, bf16* __restrict__ kb, bf16* __restrict__ vt) {
  const int lane = threadIdx.x & 31;
  const int row  = blockIdx.x * 8 + (threadIdx.x >> 5);   // [0, B*L*NH)
  const int head = row % NH_;
  const int bl   = row / NH_;
  const int b = bl / L_, l = bl % L_;

  const float* src = qkv + (size_t)bl * E_ + head * HD_;
  float x0 = src[lane];
  float x1 = src[lane + 32];

  if (head < H_ + KV_) {                         // q or k: rmsnorm + rope
    float ss = x0 * x0 + x1 * x1;
#pragma unroll
    for (int m = 16; m >= 1; m >>= 1) ss += __shfl_xor(ss, m, 32);
    float inv = rsqrtf(ss * (1.0f / HD_) + 1e-6f);
    const float* w = (head < H_) ? qw : kw;
    float n0 = x0 * inv * w[lane];
    float n1 = x1 * inv * w[lane + 32];

    float fr = __powf(10000.0f, -(float)(2 * lane) / (float)HD_);
    float th = (float)l * fr;
    float sn, cs;
    __sincosf(th, &sn, &cs);
    float o0 = n0 * cs - n1 * sn;
    float o1 = n0 * sn + n1 * cs;

    if (head < H_) {
      const float sc = 0.125f;                   // HD^-0.5 folded into Q
      bf16* dst = qb + ((size_t)(b * H_ + head) * L_ + l) * HD_;
      dst[lane] = (bf16)(o0 * sc);
      dst[lane + 32] = (bf16)(o1 * sc);
    } else {
      int kvh = head - H_;
      bf16* dst = kb + ((size_t)(b * KV_ + kvh) * L_ + l) * HD_;
      dst[lane] = (bf16)o0;
      dst[lane + 32] = (bf16)o1;
    }
  } else {                                       // v: transpose to [hd][l]
    int kvh = head - H_ - KV_;
    bf16* dst = vt + (size_t)(b * KV_ + kvh) * HD_ * L_;
    dst[(size_t)lane * L_ + l] = (bf16)x0;
    dst[(size_t)(lane + 32) * L_ + l] = (bf16)x1;
  }
}

/* ---------------- causal flash attention ----------------
   grid (B*H, L/64), 128 threads; each wave owns a 16-row q tile, loops key
   blocks of 32 with online softmax. Masking only in diagonal tail blocks;
   row-max via DPP butterfly; row-sum via an extra WMMA against ones. */
__global__ __launch_bounds__(128) void flash_fwd(
    const bf16* __restrict__ qb, const bf16* __restrict__ kb, const bf16* __restrict__ vt,
    bf16* __restrict__ ao) {
  __shared__ bf16 Ps[4][16 * 40];

  const int lane = threadIdx.x & 31, wid = threadIdx.x >> 5;
  const int lrow = lane & 15, lhalf = lane >> 4;
  const int bh = blockIdx.x, b = bh / H_, h = bh % H_, kvh = h / (H_ / KV_);
  const int q0 = blockIdx.y * 64 + wid * 16;

  const bf16* Qp = qb + (size_t)(b * H_ + h) * L_ * HD_;
  const bf16* Kp = kb + (size_t)(b * KV_ + kvh) * L_ * HD_;
  const bf16* Vp = vt + (size_t)(b * KV_ + kvh) * HD_ * L_;

  bf16x16 qf[2];
#pragma unroll
  for (int ks = 0; ks < 2; ++ks)
    qf[ks] = load_fragA(Qp + (size_t)(q0 + lrow) * HD_ + ks * 32, lhalf);

  bf16x16 onesf;
#pragma unroll
  for (int j = 0; j < 16; ++j) onesf[j] = (bf16)1.0f;

  f32x8 zero = {0.f,0.f,0.f,0.f,0.f,0.f,0.f,0.f};
  f32x8 o[4];
  f32x8 lsv = zero;                  // row sums (C layout, lane-replicated)
  float mr[8];
#pragma unroll
  for (int nf = 0; nf < 4; ++nf) o[nf] = zero;
#pragma unroll
  for (int i = 0; i < 8; ++i) mr[i] = -1.0e30f;

  bf16* P = Ps[wid];

  auto kblock = [&](int j0, bool masked) {
    // S = Q K^T  (16x32 tile of scores)
    f32x8 s[2] = {zero, zero};
#pragma unroll
    for (int ni = 0; ni < 2; ++ni)
#pragma unroll
      for (int ks = 0; ks < 2; ++ks) {
        bf16x16 kf = load_fragB(Kp + (size_t)(j0 + ni * 16 + lrow) * HD_ + ks * 32, lhalf);
        s[ni] = wmma_bf16(qf[ks], kf, s[ni]);
      }

    if (masked) {                                // diagonal blocks only
#pragma unroll
      for (int ni = 0; ni < 2; ++ni) {
        int kidx = j0 + ni * 16 + lrow;
#pragma unroll
        for (int i = 0; i < 8; ++i) {
          int qidx = q0 + i + 8 * lhalf;
          if (kidx > qidx) s[ni][i] += -1.0e9f;
        }
      }
    }

    // online softmax: row-max via DPP butterfly (VALU only), exp is TRANS
#pragma unroll
    for (int i = 0; i < 8; ++i) {
      float v = redmax16(fmaxf(s[0][i], s[1][i]));
      float mnew = fmaxf(mr[i], v);
      float alpha = __expf(mr[i] - mnew);
      mr[i] = mnew;
      s[0][i] = __expf(s[0][i] - mnew);
      s[1][i] = __expf(s[1][i] - mnew);
      lsv[i] *= alpha;
#pragma unroll
      for (int nf = 0; nf < 4; ++nf) o[nf][i] *= alpha;
    }

    // P: C-layout -> A-layout via per-wave LDS tile (in-order DS, wave-private)
#pragma unroll
    for (int f = 0; f < 2; ++f)
#pragma unroll
      for (int i = 0; i < 8; ++i)
        P[(i + 8 * lhalf) * 40 + f * 16 + lrow] = (bf16)s[f][i];
    asm volatile("s_wait_dscnt 0" ::: "memory");
    bf16x16 pf = load_fragA(P + lrow * 40, lhalf);

    // row sums: ls += P x ones  (replaces 32 ds_bpermute+adds per block)
    lsv = wmma_bf16(pf, onesf, lsv);

    // O += P V   (V^T stored [hd][l] so B-frags are contiguous loads)
#pragma unroll
    for (int nf = 0; nf < 4; ++nf) {
      bf16x16 vf = load_fragB(Vp + (size_t)(nf * 16 + lrow) * L_ + j0, lhalf);
      o[nf] = wmma_bf16(pf, vf, o[nf]);
    }
    asm volatile("" ::: "memory"); // keep P reads before next iter writes
  };

  const int nkb   = (q0 + 47) / 32;  // key blocks covering j <= q0+15
  const int nfull = (q0 + 1) / 32;   // blocks with j0+31 <= q0: mask-free

  for (int jb = 0; jb < nfull; ++jb) {
    // speculative prefetch of the next K block (global_prefetch_b8)
    __builtin_prefetch(Kp + (size_t)(jb * 32 + 32 + lane) * HD_, 0, 1);
    kblock(jb * 32, false);
  }
  for (int jb = nfull; jb < nkb; ++jb)
    kblock(jb * 32, true);

  // normalize and emit attention output; i*O_ and nf*16 fold to store offsets
  bf16* ob = ao + (size_t)(b * L_ + q0 + 8 * lhalf) * O_ + h * HD_ + lrow;
#pragma unroll
  for (int i = 0; i < 8; ++i) {
    float r = 1.0f / lsv[i];
#pragma unroll
    for (int nf = 0; nf < 4; ++nf)
      ob[i * O_ + nf * 16] = (bf16)(o[nf][i] * r);
  }
}

extern "C" void kernel_launch(void* const* d_in, const int* in_sizes, int n_in,
                              void* d_out, int out_size, void* d_ws, size_t ws_size,
                              hipStream_t stream) {
  const float* x    = (const float*)d_in[0];
  const float* Wqkv = (const float*)d_in[1];
  const float* Wout = (const float*)d_in[2];
  const float* qw   = (const float*)d_in[3];
  const float* kw   = (const float*)d_in[4];
  // d_in[5] = mask: causal, applied analytically in flash_fwd

  char* ws = (char*)d_ws;
  size_t off = 0;
  auto carve = [&](size_t bytes) -> void* {
    void* p = ws + off;
    off += (bytes + 255) & ~(size_t)255;
    return p;
  };
  bf16*  xb   = (bf16*) carve((size_t)B_ * L_ * D_ * 2);     // x in bf16
  bf16*  wqb  = (bf16*) carve((size_t)E_ * D_ * 2);          // W_qkv bf16
  bf16*  wob  = (bf16*) carve((size_t)D_ * O_ * 2);          // W_out bf16
  float* qkv  = (float*)carve((size_t)B_ * L_ * E_ * 4);     // qkv f32
  bf16*  qbf  = (bf16*) carve((size_t)B_ * H_  * L_ * HD_ * 2);
  bf16*  kbf  = (bf16*) carve((size_t)B_ * KV_ * L_ * HD_ * 2);
  bf16*  vtb  = (bf16*) carve((size_t)B_ * KV_ * HD_ * L_ * 2);
  bf16*  ao   = (bf16*) carve((size_t)B_ * L_ * O_ * 2);     // attention out bf16

  const int nx = B_ * L_ * D_, nwq = E_ * D_, nwo = D_ * O_;
  cvt_f32_bf16<<<(nx / 4 + 255) / 256, 256, 0, stream>>>(x,    xb,  nx / 4);
  cvt_f32_bf16<<<(nwq / 4 + 255) / 256, 256, 0, stream>>>(Wqkv, wqb, nwq / 4);
  cvt_f32_bf16<<<(nwo / 4 + 255) / 256, 256, 0, stream>>>(Wout, wob, nwo / 4);

  // qkv = x @ W_qkv^T : M=4096, N=1536, K=2048
  gemm_nt<E_, D_><<<dim3((B_ * L_) / 128, E_ / 128), 256, 0, stream>>>(xb, wqb, qkv);

  qkv_post<<<(B_ * L_ * NH_) / 8, 256, 0, stream>>>(qkv, qw, kw, qbf, kbf, vtb);

  flash_fwd<<<dim3(B_ * H_, L_ / 64), 128, 0, stream>>>(qbf, kbf, vtb, ao);

  // out = ao @ W_out^T : M=4096, N=2048, K=1024
  gemm_nt<D_, O_><<<dim3((B_ * L_) / 128, D_ / 128), 256, 0, stream>>>(
      ao, wob, (float*)d_out);
}
